// Attention1D_26688926778019
// MI455X (gfx1250) — matrix operand order repaired and checked by reading, not compiled, not run on hardware
//
#include <hip/hip_runtime.h>

// ---------- types ----------
typedef __bf16 bf16_t;
typedef bf16_t v16bf __attribute__((ext_vector_type(16)));
typedef bf16_t v8bf  __attribute__((ext_vector_type(8)));
typedef bf16_t v4bf  __attribute__((ext_vector_type(4)));
typedef float  v8f   __attribute__((ext_vector_type(8)));

union FragU { struct { v8bf lo, hi; } p; v16bf v; };

// ---------- geometry ----------
#define DIMM     256
#define NTOK     8
#define NHEADS   8
#define DHEAD    32
#define WGITEMS  16           // batch items per workgroup
#define ROWS     128          // WGITEMS * NTOK token-rows per WG
#define THREADS  256          // 8 waves of 32
#define RTILES   8            // 16-row tiles per WG
#define XSTR     264          // xn row stride (bf16): 256 + 8 pad -> bank stride 4
#define QSTR     776          // qkv row stride (bf16): 768 + 8 pad
#define OSTR     260          // f32 staging row stride (floats): 256 + 4 pad

// ---------- weight prep: fp32 [K, N] -> bf16 transposed [N, K] ----------
__global__ void cvt_transpose_bf16(const float* __restrict__ w,
                                   bf16_t* __restrict__ o,
                                   int K, int Ncols) {
  int t = blockIdx.x * blockDim.x + threadIdx.x;
  if (t >= K * Ncols) return;
  int n = t / K;
  int k = t - n * K;
  o[t] = (bf16_t)w[(size_t)k * Ncols + n];
}

// ---------- fused LN + QKV + attention + out-proj ----------
__global__ __launch_bounds__(THREADS)
void attn_fused(const float* __restrict__ x,
                const float* __restrict__ lnw,
                const float* __restrict__ lnb,
                const float* __restrict__ relTab,   // [14, 8]
                const int*   __restrict__ relIdx,   // [8, 8]
                const bf16_t* __restrict__ wqkvT,   // [768, 256] bf16
                const bf16_t* __restrict__ woutT,   // [256, 256] bf16
                float* __restrict__ out) {
  __shared__ bf16_t s_xn[ROWS * XSTR];    // xn bf16, reused as attn-out
  __shared__ bf16_t s_qkv[ROWS * QSTR];   // qkv bf16, reused as f32 staging
  __shared__ float  s_bias[NHEADS * NTOK * NTOK];

  const int tid  = threadIdx.x;
  const int wave = tid >> 5;
  const int lane = tid & 31;
  const int wg   = blockIdx.x;

  // ---- rel-pos bias table -> LDS as [h][i][j] ----
  for (int e = tid; e < NHEADS * NTOK * NTOK; e += THREADS) {
    int h = e >> 6, ij = e & 63;
    s_bias[e] = relTab[(size_t)relIdx[ij] * NHEADS + h];
  }

  // ---- Stage 1: LayerNorm (wave handles rows wave*16 .. +15) ----
  {
    const float4* x4 = (const float4*)x;
    const float4* w4 = (const float4*)lnw;
    const float4* b4 = (const float4*)lnb;
    float4 wa = w4[lane], wb = w4[32 + lane];
    float4 ba = b4[lane], bb = b4[32 + lane];
    for (int r = 0; r < 16; ++r) {
      int row = wave * 16 + r;
      size_t gb = ((size_t)wg * ROWS + row) * (DIMM / 4);
      float4 a = x4[gb + lane];          // cols 4*lane .. +3
      float4 b = x4[gb + 32 + lane];     // cols 128+4*lane .. +3
      float s  = a.x + a.y + a.z + a.w + b.x + b.y + b.z + b.w;
      float sq = a.x*a.x + a.y*a.y + a.z*a.z + a.w*a.w
               + b.x*b.x + b.y*b.y + b.z*b.z + b.w*b.w;
#pragma unroll
      for (int m = 16; m >= 1; m >>= 1) {
        s  += __shfl_xor(s,  m, 32);
        sq += __shfl_xor(sq, m, 32);
      }
      float mu  = s * (1.0f / 256.0f);
      float var = sq * (1.0f / 256.0f) - mu * mu;
      float ri  = rsqrtf(var + 1e-5f);
      v4bf oa = { (bf16_t)((a.x - mu) * ri * wa.x + ba.x),
                  (bf16_t)((a.y - mu) * ri * wa.y + ba.y),
                  (bf16_t)((a.z - mu) * ri * wa.z + ba.z),
                  (bf16_t)((a.w - mu) * ri * wa.w + ba.w) };
      v4bf ob = { (bf16_t)((b.x - mu) * ri * wb.x + bb.x),
                  (bf16_t)((b.y - mu) * ri * wb.y + bb.y),
                  (bf16_t)((b.z - mu) * ri * wb.z + bb.z),
                  (bf16_t)((b.w - mu) * ri * wb.w + bb.w) };
      *(v4bf*)&s_xn[row * XSTR + 4 * lane]       = oa;
      *(v4bf*)&s_xn[row * XSTR + 128 + 4 * lane] = ob;
    }
  }
  __syncthreads();

  const int lm   = lane & 15;
  const int hilo = lane >> 4;                    // 0 or 1
  // A-frag (ISA 16-bit A 16x32): lane<16 K{0..7,16..23}; lane>=16 K{8..15,24..31}
  const int aoff = lm * XSTR + hilo * 8;

  // ---- Stage 2: QKV GEMM  [128,256] x [256,768] -> bf16 qkv in LDS ----
  // Column tiles distributed over waves: each B fragment fetched ONCE per WG.
  // Software pipelined: B double-buffered across k-steps; all 8 A pairs
  // batch-issued before the 8 WMMAs so waits are graduated, not dscnt==0.
  for (int ci = 0; ci < 6; ++ci) {
    const int ct = wave * 6 + ci;                // 48 col tiles total
    v8f acc[RTILES];
#pragma unroll
    for (int rt = 0; rt < RTILES; ++rt) acc[rt] = (v8f){};
    const bf16_t* wp = wqkvT + ((size_t)(ct * 16 + lm)) * DIMM + hilo * 8;
    FragU Bb[2];
    Bb[0].p.lo = *(const v8bf*)(wp);
    Bb[0].p.hi = *(const v8bf*)(wp + 16);
#pragma unroll
    for (int kk = 0; kk < 8; ++kk) {
      // opaque (always-zero) index: keeps LDS loads un-hoistable (no spills)
      // while the base stays addrspace(3) -> ds_load_b128.
      unsigned xoff = 0;
      asm volatile("" : "+v"(xoff));
      if (kk < 7) {                              // prefetch next B a step early
        Bb[(kk + 1) & 1].p.lo = *(const v8bf*)(wp + (kk + 1) * 32);
        Bb[(kk + 1) & 1].p.hi = *(const v8bf*)(wp + (kk + 1) * 32 + 16);
      }
      FragU A[RTILES];
#pragma unroll
      for (int rt = 0; rt < RTILES; ++rt) {
        const bf16_t* ap = &s_xn[xoff + rt * 16 * XSTR + aoff + kk * 32];
        A[rt].p.lo = *(const v8bf*)(ap);
        A[rt].p.hi = *(const v8bf*)(ap + 16);
      }
#pragma unroll
      for (int rt = 0; rt < RTILES; ++rt)
        acc[rt] = __builtin_amdgcn_wmma_f32_16x16x32_bf16(
            false, A[rt].v, false, Bb[kk & 1].v, (short)0, acc[rt], false, false);
    }
    const int col = ct * 16 + lm;
#pragma unroll
    for (int rt = 0; rt < RTILES; ++rt) {
      const int rb = rt * 16 + hilo * 8;         // C: lanes>=16 hold M=8..15
#pragma unroll
      for (int r = 0; r < 8; ++r)
        s_qkv[(rb + r) * QSTR + col] = (bf16_t)acc[rt][r];
    }
  }
  __syncthreads();

  // ---- Stage 3: attention (wave-local: wave's 16 rows = its 2 items) ----
  {
    const int p    = tid >> 1;          // 128 (item, head) pairs
    const int item = p >> 3;
    const int head = p & 7;
    const int i0   = (tid & 1) * 4;     // each thread: 4 query rows
    const float scale = 0.17677669529663687f;   // 32^-0.5
    const bf16_t* qb = s_qkv + (size_t)item * NTOK * QSTR + head * DHEAD;
    for (int i = i0; i < i0 + 4; ++i) {
      float q[32];
#pragma unroll
      for (int c = 0; c < 32; ++c) q[c] = (float)qb[i * QSTR + c];
      float sv[8];
#pragma unroll
      for (int j = 0; j < 8; ++j) {
        const bf16_t* kb = qb + j * QSTR + 256;
        float a0 = 0.f, a1 = 0.f;
#pragma unroll
        for (int c = 0; c < 32; c += 2) {
          a0 += q[c]     * (float)kb[c];
          a1 += q[c + 1] * (float)kb[c + 1];
        }
        sv[j] = (a0 + a1) * scale + s_bias[head * 64 + i * 8 + j];
      }
      float m = sv[0];
#pragma unroll
      for (int j = 1; j < 8; ++j) m = fmaxf(m, sv[j]);
      float ssum = 0.f;
#pragma unroll
      for (int j = 0; j < 8; ++j) { sv[j] = __expf(sv[j] - m); ssum += sv[j]; }
      float inv = 1.0f / ssum;
      float o[32];
#pragma unroll
      for (int c = 0; c < 32; ++c) o[c] = 0.f;
#pragma unroll
      for (int j = 0; j < 8; ++j) {
        const bf16_t* vb = qb + j * QSTR + 512;
        float aj = sv[j];
#pragma unroll
        for (int c = 0; c < 32; ++c) o[c] += aj * (float)vb[c];
      }
      int orow = item * NTOK + i;
#pragma unroll
      for (int c = 0; c < 32; ++c)
        s_xn[orow * XSTR + head * DHEAD + c] = (bf16_t)(o[c] * inv);  // reuse xn buf
    }
  }
  __syncthreads();

  // ---- Stage 4: out projection [128,256] x [256,256] -> f32 staged ----
  {
    float* stage = (float*)s_qkv;                // reuse qkv LDS as f32 staging
    for (int ci = 0; ci < 2; ++ci) {
      const int ct = wave * 2 + ci;              // 16 col tiles total
      v8f acc[RTILES];
#pragma unroll
      for (int rt = 0; rt < RTILES; ++rt) acc[rt] = (v8f){};
      const bf16_t* wp = woutT + ((size_t)(ct * 16 + lm)) * DIMM + hilo * 8;
      FragU Bb[2];
      Bb[0].p.lo = *(const v8bf*)(wp);
      Bb[0].p.hi = *(const v8bf*)(wp + 16);
#pragma unroll
      for (int kk = 0; kk < 8; ++kk) {
        unsigned xoff = 0;
        asm volatile("" : "+v"(xoff));           // defeat LICM (see stage 2)
        if (kk < 7) {
          Bb[(kk + 1) & 1].p.lo = *(const v8bf*)(wp + (kk + 1) * 32);
          Bb[(kk + 1) & 1].p.hi = *(const v8bf*)(wp + (kk + 1) * 32 + 16);
        }
        FragU A[RTILES];
#pragma unroll
        for (int rt = 0; rt < RTILES; ++rt) {
          const bf16_t* ap = &s_xn[xoff + rt * 16 * XSTR + aoff + kk * 32];
          A[rt].p.lo = *(const v8bf*)(ap);
          A[rt].p.hi = *(const v8bf*)(ap + 16);
        }
#pragma unroll
        for (int rt = 0; rt < RTILES; ++rt)
          acc[rt] = __builtin_amdgcn_wmma_f32_16x16x32_bf16(
              false, A[rt].v, false, Bb[kk & 1].v, (short)0, acc[rt], false, false);
      }
      const int col = ct * 16 + lm;
#pragma unroll
      for (int rt = 0; rt < RTILES; ++rt) {
        const int rb = rt * 16 + hilo * 8;
#pragma unroll
        for (int r = 0; r < 8; ++r)
          stage[(rb + r) * OSTR + col] = acc[rt][r];
      }
    }
    __syncthreads();                             // staging is cross-wave

    // coalesced b128 global stores
    float4* o4 = (float4*)out;
    for (int r = 0; r < 16; ++r) {
      int row = wave * 16 + r;
      size_t gb = ((size_t)wg * ROWS + row) * (DIMM / 4);
      float4 v0 = *(const float4*)&stage[row * OSTR + 4 * lane];
      float4 v1 = *(const float4*)&stage[row * OSTR + 128 + 4 * lane];
      o4[gb + lane]      = v0;
      o4[gb + 32 + lane] = v1;
    }
  }
}

extern "C" void kernel_launch(void* const* d_in, const int* in_sizes, int n_in,
                              void* d_out, int out_size, void* d_ws, size_t ws_size,
                              hipStream_t stream) {
  (void)in_sizes; (void)n_in; (void)out_size; (void)ws_size;
  const float* x    = (const float*)d_in[0];
  const float* lnw  = (const float*)d_in[1];
  const float* lnb  = (const float*)d_in[2];
  const float* wqkv = (const float*)d_in[3];   // [256, 768] f32
  const float* wout = (const float*)d_in[4];   // [256, 256] f32
  const float* rtab = (const float*)d_in[5];   // [14, 8] f32
  const int*   ridx = (const int*)d_in[6];     // [8, 8] int32

  bf16_t* wqkvT = (bf16_t*)d_ws;               // 768*256 bf16 = 384 KB
  bf16_t* woutT = wqkvT + 768 * 256;           // 256*256 bf16 = 128 KB

  cvt_transpose_bf16<<<(768 * 256 + 255) / 256, 256, 0, stream>>>(wqkv, wqkvT, 256, 768);
  cvt_transpose_bf16<<<(256 * 256 + 255) / 256, 256, 0, stream>>>(wout, woutT, 256, 256);

  attn_fused<<<32768 / WGITEMS, THREADS, 0, stream>>>(
      x, lnw, lnb, rtab, ridx, wqkvT, woutT, (float*)d_out);
}
